// HMMGLM_9637906612727
// MI455X (gfx1250) — matrix operand order repaired and checked by reading, compiled, not run on hardware
//
#include <hip/hip_runtime.h>
#include <hip/hip_bf16.h>

typedef __attribute__((ext_vector_type(16))) _Float16 v16h;
typedef __attribute__((ext_vector_type(8)))  float    v8f;

// Problem constants (match reference)
constexpr int kS  = 8;
constexpr int kN  = 32;
constexpr int kT  = 262144;
constexpr int kCS = 512;          // scan chunk length
constexpr int kNC = kT / kCS;     // 512 chunks
constexpr float kEPS = 1e-16f;

__device__ __forceinline__ float fastrcp(float x) {
  return __builtin_amdgcn_rcpf(x);   // v_rcp_f32, ~1 ulp
}

// ---------------------------------------------------------------------------
// Kernel 1: emission probabilities via WMMA.
// Each wave handles a 16-timestep tile: A = conv tile (16x32 f16),
// B = W[s] slice (32x16 f16), D = logits (16x16 f32) per (state, m-halftile).
// Then Poisson log-lik, neuron-sum via xor shuffles, exp + clamp.
// ---------------------------------------------------------------------------
__global__ __launch_bounds__(256) void k_emission(
    const float* __restrict__ conv,    // (T,32)
    const float* __restrict__ weight,  // (8,32,32)
    const float* __restrict__ bias,    // (32)
    const int*   __restrict__ spikes,  // (T,32)
    float*       __restrict__ em)      // (T,8)
{
  const int lane = threadIdx.x & 31;
  const int wave = threadIdx.x >> 5;
  const int tile = blockIdx.x * 8 + wave;   // 16 timesteps per wave
  const int t0   = tile * 16;
  const int half = lane >> 4;               // 0: lanes 0-15, 1: lanes 16-31
  const int l    = lane & 15;

  // ---- A operand: conv rows, f16 16x32 layout (lane=row M, VGPRs=K) ----
  const float* row = conv + (size_t)(t0 + l) * kN;
  const int kb = half * 16;
  v16h a;
#pragma unroll
  for (int j = 0; j < 4; ++j) {
    a[2*j]     = (_Float16)row[kb + 2*j];
    a[2*j + 1] = (_Float16)row[kb + 2*j + 1];
  }
#pragma unroll
  for (int j = 0; j < 4; ++j) {
    a[8 + 2*j]     = (_Float16)row[kb + 8 + 2*j];
    a[8 + 2*j + 1] = (_Float16)row[kb + 8 + 2*j + 1];
  }
  if (t0 + 16 < kT)  // prefetch next tile's conv row -> global_prefetch_b8
    __builtin_prefetch(conv + (size_t)(t0 + 16 + l) * kN, 0, 1);

  // ---- spikes for the C-tile positions this lane owns (state-independent) --
  // C tile tt: column m = l + 16*tt ; VGPR j -> timestep t0 + j + 8*half
  float xs[2][8];
#pragma unroll
  for (int tt = 0; tt < 2; ++tt) {
    const int m = l + 16 * tt;
#pragma unroll
    for (int j = 0; j < 8; ++j) {
      const int t = t0 + j + 8 * half;
      xs[tt][j] = (float)spikes[(size_t)t * kN + m];
    }
  }
  const float bm[2] = { bias[l], bias[l + 16] };

  for (int s = 0; s < kS; ++s) {
    float lg[8];
#pragma unroll
    for (int j = 0; j < 8; ++j) lg[j] = 0.0f;

#pragma unroll
    for (int tt = 0; tt < 2; ++tt) {
      const int m = l + 16 * tt;
      // B operand: B[k=n][col=m] = weight[s][m][n]; lane reads 16 contiguous n
      const float* wrow = weight + ((size_t)s * kN + m) * kN + 16 * half;
      v16h b;
#pragma unroll
      for (int j = 0; j < 8; ++j) {
        b[2*j]     = (_Float16)wrow[2*j];
        b[2*j + 1] = (_Float16)wrow[2*j + 1];
      }
      v8f cacc = {};
      cacc = __builtin_amdgcn_wmma_f32_16x16x32_f16(
          false, a, false, b, (short)0, cacc, false, false);

#pragma unroll
      for (int j = 0; j < 8; ++j) {
        const float z = cacc[j] + bm[tt];
        // log(sigmoid(z)) = -log(1+e^-z); sigmoid via v_rcp (no div expansion)
        const float u = __expf(-z);
        const float d = 1.0f + u;
        const float r = fastrcp(d);
        lg[j] += fmaf(xs[tt][j], -__logf(d), -r);      // x*log(r) - r
      }
    }
    // sum over 32 neurons = reduce over the 16 lanes of this half-wave
#pragma unroll
    for (int mask = 1; mask < 16; mask <<= 1) {
#pragma unroll
      for (int j = 0; j < 8; ++j) lg[j] += __shfl_xor(lg[j], mask, 32);
    }
    if (l == 0) {
#pragma unroll
      for (int j = 0; j < 8; ++j) {
        const float e = fmaxf(__expf(lg[j]), kEPS);
        em[(size_t)(t0 + j + 8 * half) * kS + s] = e;
      }
    }
  }
}

// ---------------------------------------------------------------------------
// Forward pass 1: per-chunk transfer matrix P_c = prod_{t=lo..hi} diag(em_t) A^T
// 8 lanes per chunk, lane j owns column j. Scale-invariant (alpha is
// renormalized), so rescale by max(em_t) each step + periodic joint rescale.
// ---------------------------------------------------------------------------
__global__ __launch_bounds__(256) void k_fchunk(
    const float* __restrict__ em, const float* __restrict__ Atm,
    float* __restrict__ P)
{
  const int lane  = threadIdx.x & 31;
  const int wave  = threadIdx.x >> 5;
  const int j     = lane & 7;
  const int chunk = (blockIdx.x * 8 + wave) * 4 + (lane >> 3);

  float A[64];                      // A[k*8+i] (uniform -> scalar regs)
#pragma unroll
  for (int q = 0; q < 64; ++q) A[q] = Atm[q];

  float p[8];
#pragma unroll
  for (int i = 0; i < 8; ++i) p[i] = (i == j) ? 1.0f : 0.0f;

  const int lo = (chunk == 0) ? 1 : chunk * kCS;
  const int hi = (chunk + 1) * kCS - 1;
  for (int t = lo; t <= hi; ++t) {
    const float* e = em + (size_t)t * kS;
    float e0[8];
#pragma unroll
    for (int i = 0; i < 8; ++i) e0[i] = e[i];
    float emx = e0[0];
#pragma unroll
    for (int i = 1; i < 8; ++i) emx = fmaxf(emx, e0[i]);
    const float inv = fastrcp(emx);

    float pn[8];
#pragma unroll
    for (int i = 0; i < 8; ++i) {
      float acc = 0.0f;
#pragma unroll
      for (int k = 0; k < 8; ++k) acc = fmaf(A[k*8 + i], p[k], acc);
      pn[i] = acc * (e0[i] * inv);
    }
#pragma unroll
    for (int i = 0; i < 8; ++i) p[i] = pn[i];

    if ((t & 63) == 63) {           // joint rescale (same factor, all columns)
      float mx = 0.0f;
#pragma unroll
      for (int i = 0; i < 8; ++i) mx = fmaxf(mx, fabsf(p[i]));
#pragma unroll
      for (int m = 1; m < 8; m <<= 1) mx = fmaxf(mx, __shfl_xor(mx, m, 32));
      const float sc = fastrcp(fmaxf(mx, 1e-30f));
#pragma unroll
      for (int i = 0; i < 8; ++i) p[i] *= sc;
    }
  }
#pragma unroll
  for (int i = 0; i < 8; ++i) P[(size_t)chunk * 64 + i * 8 + j] = p[i];
}

// ---------------------------------------------------------------------------
// Forward pass 2: sequential scan over chunk matrices (one wave).
// bound[c] = normalized alpha at t = lo_c - 1. Also emits alpha[0], c[0].
// ---------------------------------------------------------------------------
__global__ void k_fscan(const float* __restrict__ em,
                        const float* __restrict__ P,
                        float* __restrict__ bound,
                        float* __restrict__ alpha,
                        float* __restrict__ cvec)
{
  const int lane = threadIdx.x & 31;
  const int i = lane & 7;

  float a0 = 0.125f * em[i];                 // init_p * em[0]
  float s = a0;
  s += __shfl_xor(s, 1, 32); s += __shfl_xor(s, 2, 32); s += __shfl_xor(s, 4, 32);
  float sv = a0 * fastrcp(s);
  if (lane < 8) {
    alpha[i] = sv;
    bound[i] = sv;
    if (i == 0) cvec[0] = s;
  }
  for (int c = 0; c < kNC - 1; ++c) {
    const float* Pr = P + (size_t)c * 64 + i * 8;   // row i of P_c
    float acc = 0.0f;
#pragma unroll
    for (int k = 0; k < 8; ++k) acc = fmaf(Pr[k], __shfl(sv, k, 8), acc);
    float ss = acc;
    ss += __shfl_xor(ss, 1, 32); ss += __shfl_xor(ss, 2, 32); ss += __shfl_xor(ss, 4, 32);
    sv = acc * fastrcp(ss);
    if (lane < 8) bound[(size_t)(c + 1) * 8 + i] = sv;
  }
}

// ---------------------------------------------------------------------------
// Forward pass 3: replay chunk interiors, emit alpha[t] and c[t].
// ---------------------------------------------------------------------------
__global__ __launch_bounds__(256) void k_finterior(
    const float* __restrict__ em, const float* __restrict__ Atm,
    const float* __restrict__ bound,
    float* __restrict__ alpha, float* __restrict__ cvec)
{
  const int lane  = threadIdx.x & 31;
  const int wave  = threadIdx.x >> 5;
  const int i     = lane & 7;
  const int chunk = (blockIdx.x * 8 + wave) * 4 + (lane >> 3);

  float Ac[8];                       // column i of A: A[k][i]
#pragma unroll
  for (int k = 0; k < 8; ++k) Ac[k] = Atm[k * 8 + i];

  float a = bound[(size_t)chunk * 8 + i];
  const int lo = (chunk == 0) ? 1 : chunk * kCS;
  const int hi = (chunk + 1) * kCS - 1;
  for (int t = lo; t <= hi; ++t) {
    float acc = 0.0f;
#pragma unroll
    for (int k = 0; k < 8; ++k) acc = fmaf(Ac[k], __shfl(a, k, 8), acc);
    acc *= em[(size_t)t * kS + i];
    float ct = acc;
    ct += __shfl_xor(ct, 1, 32); ct += __shfl_xor(ct, 2, 32); ct += __shfl_xor(ct, 4, 32);
    a = acc * fastrcp(ct);
    alpha[(size_t)t * kS + i] = a;
    if (i == 0) cvec[t] = ct;
  }
}

// ---------------------------------------------------------------------------
// Backward pass 1: R_c = N_lo ... N_hi with N_t = A diag(em_{t+1}) / c_{t+1}.
// No rescaling allowed: beta's absolute scale matters, and 1/c keeps it O(1).
// ---------------------------------------------------------------------------
__global__ __launch_bounds__(256) void k_bchunk(
    const float* __restrict__ em, const float* __restrict__ cvec,
    const float* __restrict__ Atm, float* __restrict__ R)
{
  const int lane  = threadIdx.x & 31;
  const int wave  = threadIdx.x >> 5;
  const int j     = lane & 7;
  const int chunk = (blockIdx.x * 8 + wave) * 4 + (lane >> 3);

  float A[64];                      // A[i*8+k]
#pragma unroll
  for (int q = 0; q < 64; ++q) A[q] = Atm[q];

  float r[8];
#pragma unroll
  for (int i = 0; i < 8; ++i) r[i] = (i == j) ? 1.0f : 0.0f;

  const int lo = chunk * kCS;
  const int hi = (chunk == kNC - 1) ? (kT - 2) : ((chunk + 1) * kCS - 1);
  for (int t = hi; t >= lo; --t) {
    const float rc = fastrcp(cvec[t + 1]);
    const float* e1 = em + (size_t)(t + 1) * kS;
    float q[8];
#pragma unroll
    for (int k = 0; k < 8; ++k) q[k] = e1[k] * rc * r[k];
    float rn[8];
#pragma unroll
    for (int i = 0; i < 8; ++i) {
      float acc = 0.0f;
#pragma unroll
      for (int k = 0; k < 8; ++k) acc = fmaf(A[i*8 + k], q[k], acc);
      rn[i] = acc;
    }
#pragma unroll
    for (int i = 0; i < 8; ++i) r[i] = rn[i];
  }
#pragma unroll
  for (int i = 0; i < 8; ++i) R[(size_t)chunk * 64 + i * 8 + j] = r[i];
}

// ---------------------------------------------------------------------------
// Backward pass 2: right-to-left scan of chunk matrices (one wave).
// ubound[c] = beta at t = hi_c + 1 (start vector of chunk c).
// ---------------------------------------------------------------------------
__global__ void k_bscan(const float* __restrict__ R, float* __restrict__ ubound)
{
  const int lane = threadIdx.x & 31;
  const int i = lane & 7;
  float u = 1.0f;
  if (lane < 8) ubound[(size_t)(kNC - 1) * 8 + i] = u;
  for (int c = kNC - 1; c >= 1; --c) {
    const float* Rr = R + (size_t)c * 64 + i * 8;
    float acc = 0.0f;
#pragma unroll
    for (int k = 0; k < 8; ++k) acc = fmaf(Rr[k], __shfl(u, k, 8), acc);
    u = acc;
    if (lane < 8) ubound[(size_t)(c - 1) * 8 + i] = u;
  }
}

// ---------------------------------------------------------------------------
// Backward pass 3: replay chunk interiors, emit beta[t].
// ---------------------------------------------------------------------------
__global__ __launch_bounds__(256) void k_binterior(
    const float* __restrict__ em, const float* __restrict__ cvec,
    const float* __restrict__ Atm, const float* __restrict__ ubound,
    float* __restrict__ beta)
{
  const int lane  = threadIdx.x & 31;
  const int wave  = threadIdx.x >> 5;
  const int i     = lane & 7;
  const int chunk = (blockIdx.x * 8 + wave) * 4 + (lane >> 3);

  float Ar[8];                       // row i of A: A[i][k]
#pragma unroll
  for (int k = 0; k < 8; ++k) Ar[k] = Atm[i * 8 + k];

  float b = ubound[(size_t)chunk * 8 + i];
  const int lo = chunk * kCS;
  const int hi = (chunk == kNC - 1) ? (kT - 2) : ((chunk + 1) * kCS - 1);
  if (chunk == kNC - 1) beta[(size_t)(kT - 1) * kS + i] = 1.0f;

  for (int t = hi; t >= lo; --t) {
    const float q  = em[(size_t)(t + 1) * kS + i] * b;   // em_{t+1}[i]*b[i]
    const float rc = fastrcp(cvec[t + 1]);
    float acc = 0.0f;
#pragma unroll
    for (int k = 0; k < 8; ++k) acc = fmaf(Ar[k], __shfl(q, k, 8), acc);
    b = acc * rc;
    beta[(size_t)t * kS + i] = b;
  }
}

// ---------------------------------------------------------------------------
// Outputs: gamma = alpha * beta ; xi[t-1,i,j] = a[t-1,i]*em[t,j]*A[i,j]*b[t,j]/c[t]
// ---------------------------------------------------------------------------
__global__ void k_gamma(const float* __restrict__ alpha,
                        const float* __restrict__ beta,
                        float* __restrict__ out, int n)
{
  const int idx = blockIdx.x * blockDim.x + threadIdx.x;
  if (idx < n) out[idx] = alpha[idx] * beta[idx];
}

__global__ void k_xi(const float* __restrict__ alpha,
                     const float* __restrict__ beta,
                     const float* __restrict__ em,
                     const float* __restrict__ cvec,
                     const float* __restrict__ Atm,
                     float* __restrict__ out)
{
  const size_t idx = (size_t)blockIdx.x * blockDim.x + threadIdx.x;
  const size_t total = (size_t)(kT - 1) * 64;
  if (idx >= total) return;
  const int    e  = (int)(idx & 63);
  const size_t tq = idx >> 6;             // t-1
  const int i = e >> 3, j = e & 7;
  const size_t t = tq + 1;
  out[idx] = alpha[tq * kS + i] * em[t * kS + j] * Atm[e] *
             beta[t * kS + j] * fastrcp(cvec[t]);
}

// ---------------------------------------------------------------------------
extern "C" void kernel_launch(void* const* d_in, const int* in_sizes, int n_in,
                              void* d_out, int out_size, void* d_ws, size_t ws_size,
                              hipStream_t stream) {
  (void)in_sizes; (void)n_in; (void)out_size; (void)ws_size;
  const int*   spikes = (const int*)  d_in[0];   // (T,32) int32
  const float* conv   = (const float*)d_in[1];   // (T,32)
  const float* weight = (const float*)d_in[2];   // (8,32,32)
  const float* bias   = (const float*)d_in[3];   // (32)
  const float* Atm    = (const float*)d_in[4];   // (8,8)

  // Workspace layout (floats): ~26.5 MB total
  float* em    = (float*)d_ws;                       // T*8
  float* alpha = em    + (size_t)kT * kS;            // T*8
  float* beta  = alpha + (size_t)kT * kS;            // T*8
  float* cvec  = beta  + (size_t)kT * kS;            // T
  float* P     = cvec  + (size_t)kT;                 // NC*64
  float* Rm    = P     + (size_t)kNC * 64;           // NC*64
  float* sb    = Rm    + (size_t)kNC * 64;           // NC*8
  float* ub    = sb    + (size_t)kNC * 8;            // NC*8

  float* out_gamma = (float*)d_out;                  // T*8
  float* out_xi    = out_gamma + (size_t)kT * kS;    // (T-1)*64

  // 1) emission via WMMA: 16 timesteps per wave, 8 waves per block
  k_emission<<<kT / (16 * 8), 256, 0, stream>>>(conv, weight, bias, spikes, em);

  // 2-4) forward: chunk products -> boundary scan -> interior replay
  k_fchunk   <<<kNC / 32, 256, 0, stream>>>(em, Atm, P);
  k_fscan    <<<1, 32, 0, stream>>>(em, P, sb, alpha, cvec);
  k_finterior<<<kNC / 32, 256, 0, stream>>>(em, Atm, sb, alpha, cvec);

  // 5-7) backward: chunk products -> boundary scan -> interior replay
  k_bchunk   <<<kNC / 32, 256, 0, stream>>>(em, cvec, Atm, Rm);
  k_bscan    <<<1, 32, 0, stream>>>(Rm, ub);
  k_binterior<<<kNC / 32, 256, 0, stream>>>(em, cvec, Atm, ub, beta);

  // 8-9) outputs
  const int ngamma = kT * kS;
  k_gamma<<<(ngamma + 255) / 256, 256, 0, stream>>>(alpha, beta, out_gamma, ngamma);
  const size_t nxi = (size_t)(kT - 1) * 64;
  k_xi<<<(unsigned)((nxi + 255) / 256), 256, 0, stream>>>(alpha, beta, em, cvec,
                                                          Atm, out_xi);
}
